// GPT_33526514713119
// MI455X (gfx1250) — matrix operand order repaired and checked by reading, not compiled, not run on hardware
//
#include <hip/hip_runtime.h>
#include <hip/hip_bf16.h>
#include <math.h>

// ---------------------------------------------------------------------------
// GPT-2 small forward (B=4, T=1024, C=768, H=12, L=6, V=50257) for gfx1250.
// All GEMMs via v_wmma_f32_16x16x32_f16 (wave32, 16x16x32 tiles).
// Weights pre-converted to f16 in workspace (gated on ws_size, f32 fallback).
// Fragments loaded directly from global (192MB L2 covers the working set);
// attention softmax staged through LDS (ds_load_b128 for P fragments).
// ---------------------------------------------------------------------------

typedef __attribute__((ext_vector_type(16))) _Float16 v16h;
typedef __attribute__((ext_vector_type(8)))  _Float16 v8h;
typedef __attribute__((ext_vector_type(4)))  _Float16 v4h;
typedef __attribute__((ext_vector_type(8)))  float    v8f;
typedef __attribute__((ext_vector_type(4)))  float    v4f;

#define GPT_B 4
#define GPT_T 1024
#define GPT_C 768
#define GPT_H 12
#define GPT_L 6
#define GPT_V 50257
#define GPT_M (GPT_B * GPT_T)   // 4096 rows
#define GPT_3C (3 * GPT_C)      // 2304

__device__ __forceinline__ v8f wmma16(v16h a, v16h b, v8f c) {
    // D = A(16x32 f16) * B(32x16 f16) + C(16x16 f32)
    return __builtin_amdgcn_wmma_f32_16x16x32_f16(
        /*neg_a=*/false, a, /*neg_b=*/false, b,
        /*c_mod=*/(short)0, c, /*reuse_a=*/false, /*reuse_b=*/false);
}

__device__ __forceinline__ v16h make_v16(v8h lo, v8h hi) {
    v16h o;
#pragma unroll
    for (int i = 0; i < 8; i++) { o[i] = lo[i]; o[i + 8] = hi[i]; }
    return o;
}

// A fragment (16x32, f16 row-major source, row stride lda halfs).
// lane<16: row=lane, K = k..k+15
// lane>=16: same row, K = k+16..k+31
__device__ __forceinline__ v16h load_a_h(const _Float16* base, int lda, int k, int lane) {
    const _Float16* p = base + (size_t)(lane & 15) * lda + k + ((lane >> 4) << 4);
    return make_v16(*(const v8h*)p, *(const v8h*)(p + 8));
}

// A fragment from f32 source (LDS softmax probs), converted to f16.
__device__ __forceinline__ v16h load_a_f32(const float* base, int lda, int k, int lane) {
    const float* p = base + (size_t)(lane & 15) * lda + k + ((lane >> 4) << 4);
    v4f a0 = *(const v4f*)p, a1 = *(const v4f*)(p + 4);
    v4f b0 = *(const v4f*)(p + 8), b1 = *(const v4f*)(p + 12);
    v16h o;
#pragma unroll
    for (int i = 0; i < 4; i++) {
        o[i]      = (_Float16)a0[i]; o[i + 4]  = (_Float16)a1[i];
        o[i + 8]  = (_Float16)b0[i]; o[i + 12] = (_Float16)b1[i];
    }
    return o;
}

// B fragment (32x16): logical column n stored as contiguous row n of an [N,K]
// row-major matrix. lane<16: K=k..k+15, col=lane; lane>=16: K=k+16..k+31.
__device__ __forceinline__ v16h load_bT_h(const _Float16* base, int ldb, int k, int lane) {
    const _Float16* p = base + (size_t)(lane & 15) * ldb + k + ((lane >> 4) << 4);
    return make_v16(*(const v8h*)p, *(const v8h*)(p + 8));
}

// ---------------------------------------------------------------------------
// f32 -> f16 weight pre-conversion (one streaming pass, amortized).
// ---------------------------------------------------------------------------
__global__ __launch_bounds__(256) void cvt_kernel(
    const float* __restrict__ src, _Float16* __restrict__ dst, int n4) {
    int i = blockIdx.x * 256 + threadIdx.x;   // one float4 per thread
    if (i >= n4) return;
    v4f v = *(const v4f*)(src + (size_t)i * 4);
    v4h o;
#pragma unroll
    for (int j = 0; j < 4; j++) o[j] = (_Float16)v[j];
    *(v4h*)(dst + (size_t)i * 4) = o;
}

// ---------------------------------------------------------------------------
// Embedding: x[b*T+t, :] = wte[idx[b,t], :] + wpe[t, :]
// ---------------------------------------------------------------------------
__global__ __launch_bounds__(256) void embed_kernel(
    const int* __restrict__ idx, const float* __restrict__ wte,
    const float* __restrict__ wpe, float* __restrict__ x) {
    int row = blockIdx.x;
    int t = row & (GPT_T - 1);
    int tok = idx[row];
#pragma unroll
    for (int i = 0; i < 3; i++) {
        int c = threadIdx.x + i * 256;
        x[(size_t)row * GPT_C + c] =
            wte[(size_t)tok * GPT_C + c] + wpe[(size_t)t * GPT_C + c];
    }
}

// ---------------------------------------------------------------------------
// LayerNorm over C=768, output f16 (feeds WMMA GEMMs). One block per row.
// ---------------------------------------------------------------------------
__global__ __launch_bounds__(256) void ln_kernel(
    const float* __restrict__ x, const float* __restrict__ w,
    const float* __restrict__ b, _Float16* __restrict__ out) {
    __shared__ float rs[256], rq[256];
    int row = blockIdx.x, tid = threadIdx.x;
    const float* xr = x + (size_t)row * GPT_C;
    float v0 = xr[tid], v1 = xr[tid + 256], v2 = xr[tid + 512];
    rs[tid] = v0 + v1 + v2;
    rq[tid] = v0 * v0 + v1 * v1 + v2 * v2;
    __syncthreads();
    for (int off = 128; off > 0; off >>= 1) {
        if (tid < off) { rs[tid] += rs[tid + off]; rq[tid] += rq[tid + off]; }
        __syncthreads();
    }
    float mean = rs[0] * (1.0f / GPT_C);
    float var  = rq[0] * (1.0f / GPT_C) - mean * mean;
    float inv  = rsqrtf(var + 1e-5f);
    _Float16* orow = out + (size_t)row * GPT_C;
    orow[tid]       = (_Float16)((v0 - mean) * inv * w[tid]       + b[tid]);
    orow[tid + 256] = (_Float16)((v1 - mean) * inv * w[tid + 256] + b[tid + 256]);
    orow[tid + 512] = (_Float16)((v2 - mean) * inv * w[tid + 512] + b[tid + 512]);
}

// ---------------------------------------------------------------------------
// GEMM: out[M,N] = A_f16[M,K] @ W[N,K]^T (+bias)(+residual)(+gelu)
// Block = 256 threads = 8 waves; wave w -> rows [(by*8+w)*16, +16),
// cols [bx*64, +64) as 4 WMMA accumulators (A fragment reused 4x).
// WF16: weights are pre-converted f16 (no inner-loop cvt); else f32 + cvt.
// Only EPI_F32_PLAIN (logits, ragged N=50257) pays clamp/guard compares.
// ---------------------------------------------------------------------------
enum { EPI_F16_BIAS = 0, EPI_F32_RES = 1, EPI_F16_GELU = 2, EPI_F32_PLAIN = 3 };

template <int EPI, bool WF16>
__global__ __launch_bounds__(256) void gemm_kernel(
    const _Float16* __restrict__ A,
    const _Float16* __restrict__ Wh, const float* __restrict__ Wf,
    const float* __restrict__ bias, const float* __restrict__ res,
    _Float16* __restrict__ outH, float* __restrict__ outF,
    int M, int N, int K) {
    int lane = threadIdx.x & 31;
    int wave = threadIdx.x >> 5;
    int m0 = (blockIdx.y * 8 + wave) * 16;
    int n0 = blockIdx.x * 64;
    int r  = lane & 15;
    int hi = lane >> 4;

    v8f acc[4] = {{0}, {0}, {0}, {0}};
    const _Float16* Abase = A + (size_t)(m0 + r) * K + (hi << 4);

    for (int k = 0; k < K; k += 32) {
        // prefetch next k-step's A line (emits global_prefetch_b8)
        if (k + 32 < K) __builtin_prefetch(Abase + k + 32, 0, 3);
        v16h af = make_v16(*(const v8h*)(Abase + k), *(const v8h*)(Abase + k + 8));
#pragma unroll
        for (int j = 0; j < 4; j++) {
            int n = n0 + j * 16 + r;
            if (EPI == EPI_F32_PLAIN) { if (n >= N) n = N - 1; }  // ragged edge
            v16h bf;
            if (WF16) {
                const _Float16* wp = Wh + (size_t)n * K + k + (hi << 4);
                if (k + 32 < K) __builtin_prefetch(wp + 32, 0, 3);
                bf = make_v16(*(const v8h*)wp, *(const v8h*)(wp + 8));
            } else {
                const float* wp = Wf + (size_t)n * K + k + (hi << 4);
                if (k + 32 < K) __builtin_prefetch(wp + 32, 0, 3);
                v4f w0 = *(const v4f*)wp,       w1 = *(const v4f*)(wp + 4);
                v4f w2 = *(const v4f*)(wp + 8), w3 = *(const v4f*)(wp + 12);
#pragma unroll
                for (int i = 0; i < 4; i++) {
                    bf[i]      = (_Float16)w0[i]; bf[i + 4]  = (_Float16)w1[i];
                    bf[i + 8]  = (_Float16)w2[i]; bf[i + 12] = (_Float16)w3[i];
                }
            }
            acc[j] = wmma16(af, bf, acc[j]);
        }
    }

    // Epilogue: C layout -> lane col = n0+16j+(lane&15), row = m0+rr+8*hi
#pragma unroll
    for (int j = 0; j < 4; j++) {
        int n = n0 + j * 16 + r;
        if (EPI == EPI_F32_PLAIN) { if (n >= N) continue; }
        float bv = (EPI == EPI_F32_PLAIN) ? 0.0f : bias[n];
#pragma unroll
        for (int rr = 0; rr < 8; rr++) {
            int m = m0 + rr + (hi << 3);
            float v = acc[j][rr] + bv;
            size_t o = (size_t)m * N + n;
            if (EPI == EPI_F16_BIAS) {
                outH[o] = (_Float16)v;
            } else if (EPI == EPI_F32_RES) {
                outF[o] = res[o] + v;
            } else if (EPI == EPI_F16_GELU) {
                outH[o] = (_Float16)(0.5f * v * (1.0f + erff(v * 0.70710678f)));
            } else {
                outF[o] = v;
            }
        }
    }
}

// ---------------------------------------------------------------------------
// Attention: one wave per (b, h, 16-row q-block). S = QK^T*scale (WMMA over
// D=64), causal mask only on the diagonal tile, LDS-resident softmax (wave32
// half-lane reductions via shfl_xor 16), then Y = P @ V with P fragments
// re-read from LDS (ds_load_b128) and converted to f16.
// ---------------------------------------------------------------------------
__global__ __launch_bounds__(32) void attn_kernel(
    const _Float16* __restrict__ qkv, _Float16* __restrict__ y) {
    extern __shared__ float S[];   // 16 x 1024 f32 scores = 64 KB
    int bid = blockIdx.x;
    int qb = bid & 63;
    int h  = (bid >> 6) % GPT_H;
    int b  = bid / (64 * GPT_H);
    int lane = threadIdx.x;
    int r = lane & 15, hi = lane >> 4;

    const int kmax = (qb + 1) * 16;
    const int kpad = (kmax + 31) & ~31;
    const float scale = 0.125f;    // 1/sqrt(64)

    const _Float16* Qb = qkv + (size_t)(b * GPT_T + qb * 16) * GPT_3C + h * 64;
    // Q fragments are invariant across all k-tiles: hoist out of the loop.
    v16h qa0 = load_a_h(Qb, GPT_3C, 0,  lane);
    v16h qa1 = load_a_h(Qb, GPT_3C, 32, lane);

    // Phase 1a: strictly-below-diagonal tiles (no mask compares in hot loop)
    for (int kb = 0; kb < qb; kb++) {
        const _Float16* Kb =
            qkv + (size_t)(b * GPT_T + kb * 16) * GPT_3C + GPT_C + h * 64;
        v8f acc = {0};
        acc = wmma16(qa0, load_bT_h(Kb, GPT_3C, 0,  lane), acc);
        acc = wmma16(qa1, load_bT_h(Kb, GPT_3C, 32, lane), acc);
        float* srow = S + kb * 16 + r;
#pragma unroll
        for (int rr = 0; rr < 8; rr++)
            srow[(rr + (hi << 3)) * GPT_T] = acc[rr] * scale;
    }
    // Phase 1b: diagonal tile with causal mask
    {
        const _Float16* Kb =
            qkv + (size_t)(b * GPT_T + qb * 16) * GPT_3C + GPT_C + h * 64;
        v8f acc = {0};
        acc = wmma16(qa0, load_bT_h(Kb, GPT_3C, 0,  lane), acc);
        acc = wmma16(qa1, load_bT_h(Kb, GPT_3C, 32, lane), acc);
        float* srow = S + qb * 16 + r;
#pragma unroll
        for (int rr = 0; rr < 8; rr++) {
            int mrow = rr + (hi << 3);
            float sv = (r <= mrow) ? acc[rr] * scale : -1e30f;
            srow[mrow * GPT_T] = sv;
        }
    }
    // zero-pad odd 16-column tail up to a 32-wide K chunk
    if (kpad > kmax) {
        for (int i = lane; i < 256; i += 32)
            S[(i >> 4) * GPT_T + kmax + (i & 15)] = 0.0f;
    }
    __syncthreads();

    // Phase 2: softmax; lanes L and L^16 split row (L&15) even/odd columns
    float mx = -1e30f;
    for (int kk = hi; kk < kmax; kk += 2) mx = fmaxf(mx, S[r * GPT_T + kk]);
    mx = fmaxf(mx, __shfl_xor(mx, 16, 32));
    float sum = 0.0f;
    for (int kk = hi; kk < kmax; kk += 2) {
        float e = __expf(S[r * GPT_T + kk] - mx);
        S[r * GPT_T + kk] = e;
        sum += e;
    }
    sum += __shfl_xor(sum, 16, 32);
    float invs = 1.0f / sum;
    for (int kk = hi; kk < kmax; kk += 2) S[r * GPT_T + kk] *= invs;
    __syncthreads();

    // Phase 3: Y = P @ V  (N = d in [0,64), 4 accumulator tiles)
    v8f yacc[4] = {{0}, {0}, {0}, {0}};
    for (int kc = 0; kc < kpad; kc += 32) {
        v16h pa = load_a_f32(S, GPT_T, kc, lane);
#pragma unroll
        for (int j = 0; j < 4; j++) {
            // B column n = d; V values along t_k are strided by 3C (L2 hits)
            const _Float16* vp = qkv +
                (size_t)(b * GPT_T + kc + (hi << 4)) * GPT_3C +
                2 * GPT_C + h * 64 + j * 16 + r;
            v16h bf;
#pragma unroll
            for (int i = 0; i < 16; i++) bf[i] = vp[(size_t)i * GPT_3C];
            yacc[j] = wmma16(pa, bf, yacc[j]);
        }
    }
#pragma unroll
    for (int j = 0; j < 4; j++) {
        int d = j * 16 + r;
#pragma unroll
        for (int rr = 0; rr < 8; rr++) {
            int t = qb * 16 + rr + (hi << 3);
            y[(size_t)(b * GPT_T + t) * GPT_C + h * 64 + d] = (_Float16)yacc[j][rr];
        }
    }
}

// ---------------------------------------------------------------------------
// Host driver
// ---------------------------------------------------------------------------
extern "C" void kernel_launch(void* const* d_in, const int* in_sizes, int n_in,
                              void* d_out, int out_size, void* d_ws, size_t ws_size,
                              hipStream_t stream) {
    (void)in_sizes; (void)n_in; (void)out_size;
    const int*   idx    = (const int*)  d_in[0];
    const float* wte    = (const float*)d_in[1];
    const float* wpe    = (const float*)d_in[2];
    const float* ln1_w  = (const float*)d_in[3];
    const float* ln1_b  = (const float*)d_in[4];
    const float* attn_w = (const float*)d_in[5];
    const float* attn_b = (const float*)d_in[6];
    const float* proj_w = (const float*)d_in[7];
    const float* proj_b = (const float*)d_in[8];
    const float* ln2_w  = (const float*)d_in[9];
    const float* ln2_b  = (const float*)d_in[10];
    const float* fc_w   = (const float*)d_in[11];
    const float* fc_b   = (const float*)d_in[12];
    const float* fc2_w  = (const float*)d_in[13];
    const float* fc2_b  = (const float*)d_in[14];
    const float* lnf_w  = (const float*)d_in[15];
    const float* lnf_b  = (const float*)d_in[16];

    const int M = GPT_M, C = GPT_C;
    char* ws = (char*)d_ws;
    size_t off = 0;
    auto alloc = [&](size_t bytes) -> void* {
        void* p = ws + off;
        off = (off + bytes + 255) & ~(size_t)255;
        return p;
    };
    // Activations (always used)
    float*    x   = (float*)   alloc((size_t)M * C * sizeof(float));
    _Float16* hbf = (_Float16*)alloc((size_t)M * C * sizeof(_Float16));
    _Float16* qkv = (_Float16*)alloc((size_t)M * 3 * C * sizeof(_Float16));
    _Float16* yat = (_Float16*)alloc((size_t)M * C * sizeof(_Float16));
    _Float16* ffb = (_Float16*)alloc((size_t)M * 4 * C * sizeof(_Float16));
    // f16 weight cache (used only if workspace is large enough)
    const size_t nAttnW = (size_t)GPT_L * 3 * C * C;
    const size_t nProjW = (size_t)GPT_L * C * C;
    const size_t nFcW   = (size_t)GPT_L * 4 * C * C;
    const size_t nFc2W  = (size_t)GPT_L * C * 4 * C;
    const size_t nWte   = (size_t)GPT_V * C;
    _Float16* attn_w16 = (_Float16*)alloc(nAttnW * sizeof(_Float16));
    _Float16* proj_w16 = (_Float16*)alloc(nProjW * sizeof(_Float16));
    _Float16* fc_w16   = (_Float16*)alloc(nFcW   * sizeof(_Float16));
    _Float16* fc2_w16  = (_Float16*)alloc(nFc2W  * sizeof(_Float16));
    _Float16* wte16    = (_Float16*)alloc(nWte   * sizeof(_Float16));
    const bool wf16 = (ws_size >= off);

    embed_kernel<<<M, 256, 0, stream>>>(idx, wte, wpe, x);

    const dim3 gQKV(GPT_3C / 64, M / 128);      // 36 x 32
    const dim3 gC  (C / 64,      M / 128);      // 12 x 32
    const dim3 gFC (4 * C / 64,  M / 128);      // 48 x 32
    const dim3 gLM ((GPT_V + 63) / 64, M / 128);
    const int  nAttnBlk = GPT_B * GPT_H * (GPT_T / 16);
    const int  attnLds  = 16 * GPT_T * sizeof(float);

    if (wf16) {
        // One streaming pass converting all weights to f16 (halves GEMM
        // weight bandwidth, removes inner-loop v_cvt from compute-bound GEMMs)
        auto cvt = [&](const float* s, _Float16* d, size_t n) {
            int n4 = (int)(n / 4);
            cvt_kernel<<<(n4 + 255) / 256, 256, 0, stream>>>(s, d, n4);
        };
        cvt(attn_w, attn_w16, nAttnW);
        cvt(proj_w, proj_w16, nProjW);
        cvt(fc_w,   fc_w16,   nFcW);
        cvt(fc2_w,  fc2_w16,  nFc2W);
        cvt(wte,    wte16,    nWte);

        for (int l = 0; l < GPT_L; l++) {
            ln_kernel<<<M, 256, 0, stream>>>(x, ln1_w + l * C, ln1_b + l * C, hbf);
            gemm_kernel<EPI_F16_BIAS, true><<<gQKV, 256, 0, stream>>>(
                hbf, attn_w16 + (size_t)l * 3 * C * C, nullptr,
                attn_b + l * 3 * C, nullptr, qkv, nullptr, M, 3 * C, C);
            attn_kernel<<<nAttnBlk, 32, attnLds, stream>>>(qkv, yat);
            gemm_kernel<EPI_F32_RES, true><<<gC, 256, 0, stream>>>(
                yat, proj_w16 + (size_t)l * C * C, nullptr,
                proj_b + l * C, x, nullptr, x, M, C, C);
            ln_kernel<<<M, 256, 0, stream>>>(x, ln2_w + l * C, ln2_b + l * C, hbf);
            gemm_kernel<EPI_F16_GELU, true><<<gFC, 256, 0, stream>>>(
                hbf, fc_w16 + (size_t)l * 4 * C * C, nullptr,
                fc_b + l * 4 * C, nullptr, ffb, nullptr, M, 4 * C, C);
            gemm_kernel<EPI_F32_RES, true><<<gC, 256, 0, stream>>>(
                ffb, fc2_w16 + (size_t)l * C * 4 * C, nullptr,
                fc2_b + l * C, x, nullptr, x, M, C, 4 * C);
        }
        ln_kernel<<<M, 256, 0, stream>>>(x, lnf_w, lnf_b, hbf);
        gemm_kernel<EPI_F32_PLAIN, true><<<gLM, 256, 0, stream>>>(
            hbf, wte16, nullptr, nullptr, nullptr, nullptr, (float*)d_out,
            M, GPT_V, C);
    } else {
        // Fallback: convert weights on the fly inside the GEMM
        for (int l = 0; l < GPT_L; l++) {
            ln_kernel<<<M, 256, 0, stream>>>(x, ln1_w + l * C, ln1_b + l * C, hbf);
            gemm_kernel<EPI_F16_BIAS, false><<<gQKV, 256, 0, stream>>>(
                hbf, nullptr, attn_w + (size_t)l * 3 * C * C,
                attn_b + l * 3 * C, nullptr, qkv, nullptr, M, 3 * C, C);
            attn_kernel<<<nAttnBlk, 32, attnLds, stream>>>(qkv, yat);
            gemm_kernel<EPI_F32_RES, false><<<gC, 256, 0, stream>>>(
                yat, nullptr, proj_w + (size_t)l * C * C,
                proj_b + l * C, x, nullptr, x, M, C, C);
            ln_kernel<<<M, 256, 0, stream>>>(x, ln2_w + l * C, ln2_b + l * C, hbf);
            gemm_kernel<EPI_F16_GELU, false><<<gFC, 256, 0, stream>>>(
                hbf, nullptr, fc_w + (size_t)l * 4 * C * C,
                fc_b + l * 4 * C, nullptr, ffb, nullptr, M, 4 * C, C);
            gemm_kernel<EPI_F32_RES, false><<<gC, 256, 0, stream>>>(
                ffb, nullptr, fc2_w + (size_t)l * C * 4 * C,
                fc2_b + l * C, x, nullptr, x, M, C, 4 * C);
        }
        ln_kernel<<<M, 256, 0, stream>>>(x, lnf_w, lnf_b, hbf);
        gemm_kernel<EPI_F32_PLAIN, false><<<gLM, 256, 0, stream>>>(
            hbf, nullptr, wte, nullptr, nullptr, nullptr, (float*)d_out,
            M, GPT_V, C);
    }
}
